// MAB_41077067219485
// MI455X (gfx1250) — matrix operand order repaired and checked by reading, not compiled, not run on hardware
//
#include <hip/hip_runtime.h>
#include <hip/hip_bf16.h>

typedef __bf16 bf16;
typedef __attribute__((ext_vector_type(16))) bf16  v16bf;
typedef __attribute__((ext_vector_type(8)))  bf16  v8bf;
typedef __attribute__((ext_vector_type(8)))  float v8f;
typedef __attribute__((ext_vector_type(4)))  unsigned int v4u;
typedef __attribute__((ext_vector_type(8)))  int   v8i;
typedef __attribute__((ext_vector_type(4)))  int   v4i;

union AFrag { v16bf v; v8bf h[2]; };

__device__ inline v8f wmma_bf16(v16bf a, v16bf b, v8f c) {
  // D = A(16x32 bf16) * B(32x16 bf16) + C(16x16 f32)
  return __builtin_amdgcn_wmma_f32_16x16x32_bf16(false, a, false, b, (short)0, c,
                                                 false, false);
}

#if __has_builtin(__builtin_amdgcn_tensor_load_to_lds) && \
    __has_builtin(__builtin_amdgcn_s_wait_tensorcnt)
#define USE_TDM 1
#endif

#ifdef USE_TDM
// 2-D tile load via Tensor Data Mover (D# groups per CDNA5 ISA ch.8):
//   tile0 elements contiguous (data_size=2B), tile1 rows of stride `stride0`
__device__ inline void tdm_load_2d(unsigned lds_off, unsigned long long gaddr,
                                   unsigned tile0, unsigned tile1,
                                   unsigned stride0) {
  v4u g0;
  g0[0] = 1u;                                        // count=1, user mode
  g0[1] = lds_off;                                   // lds_addr (bytes)
  g0[2] = (unsigned)(gaddr & 0xFFFFFFFFull);         // global_addr[31:0]
  g0[3] = (unsigned)((gaddr >> 32) & 0x1FFFFFFull)   // global_addr[56:32]
          | (2u << 30);                              // type=2 ("image")
  v8i g1;
  g1[0] = (int)(1u << 16);                           // data_size=1 -> 2 bytes
  g1[1] = (int)((tile0 & 0xFFFFu) << 16);            // tensor_dim0[15:0]=tile0
  g1[2] = (int)((tile1 & 0xFFFFu) << 16);            // tensor_dim1[15:0]=tile1
  g1[3] = (int)((tile0 & 0xFFFFu) << 16);            // tile_dim0=tile0
  g1[4] = (int)(tile1 & 0xFFFFu);                    // tile_dim1=tile1, tile_dim2=0
  g1[5] = (int)stride0;                              // tensor_dim0_stride[31:0]
  g1[6] = 0;
  g1[7] = 0;
  v4i z4 = {0, 0, 0, 0};
  v8i z8 = {0, 0, 0, 0, 0, 0, 0, 0};
  __builtin_amdgcn_tensor_load_to_lds(g0, g1, z4, z4, z8, 0);
}
#endif

// ---------------------------------------------------------------- fp32 -> bf16
__global__ void k_cvt(const float* __restrict__ in, bf16* __restrict__ out, int n) {
  int i = blockIdx.x * 256 + threadIdx.x;
  if (i < n) out[i] = (bf16)in[i];
}

// ---------------------------------------------------------------- generic GEMM
// C[m,n] = sum_k A[m,k]*W[n,k] + bias[n]; A fp32 Mx256 row-major (converted to
// bf16 inline), W 256x256 bf16 row-major (n,k). One wave = 16x64 strip.
// mode 0: q proj -> outf fp32 + outb bf16 (row-major)
// mode 1: k proj -> outb bf16 (row-major)
// mode 2: v proj -> outb bf16 transposed vT[b][d][key]
// mode 3: ffn    -> outf = res + relu(acc+bias)
__global__ __launch_bounds__(128) void k_gemm(
    const float* __restrict__ A, const bf16* __restrict__ W,
    const float* __restrict__ bias, float* __restrict__ outf,
    bf16* __restrict__ outb, const float* __restrict__ res, int mode)
{
  const int wid  = blockIdx.x * 4 + (threadIdx.x >> 5);   // 2048 waves
  const int lane = threadIdx.x & 31;
  const int lo = lane & 15, hi = lane >> 4;
  const int m0 = (wid >> 2) << 4;    // 512 m-tiles
  const int n0 = (wid & 3) << 6;     // 4 strips of 64 columns

  const float* arow = A + (size_t)(m0 + lo) * 256;
  const bf16* wrow0 = W + (size_t)(n0 + lo) * 256;

  const v8f vzero = {};
  v8f acc[4];
#pragma unroll
  for (int t = 0; t < 4; ++t) acc[t] = vzero;

#pragma unroll
  for (int kb = 0; kb < 256; kb += 32) {
    v8f f0 = *(const v8f*)(arow + kb + hi * 16);
    v8f f1 = *(const v8f*)(arow + kb + hi * 16 + 8);
    AFrag a;
#pragma unroll
    for (int i = 0; i < 8; ++i) { a.h[0][i] = (bf16)f0[i]; a.h[1][i] = (bf16)f1[i]; }
#pragma unroll
    for (int t = 0; t < 4; ++t) {
      v16bf b = *(const v16bf*)(wrow0 + (size_t)t * 16 * 256 + kb + hi * 16);
      acc[t] = wmma_bf16(a.v, b, acc[t]);
    }
  }

#pragma unroll
  for (int t = 0; t < 4; ++t) {
    const int col = n0 + t * 16 + lo;
    const float bias_n = bias[col];
#pragma unroll
    for (int r = 0; r < 8; ++r) {
      const int m = m0 + hi * 8 + r;
      const size_t idx = (size_t)m * 256 + col;
      const float v = acc[t][r] + bias_n;
      if (mode == 0)      { outf[idx] = v; outb[idx] = (bf16)v; }
      else if (mode == 1) { outb[idx] = (bf16)v; }
      else if (mode == 2) {
        const int bb = m >> 11, key = m & 2047;
        outb[(size_t)bb * (256 * 2048) + (size_t)col * 2048 + key] = (bf16)v;
      } else {
        outf[idx] = res[idx] + fmaxf(v, 0.0f);
      }
    }
  }
}

// ------------------------------------------------------------ flash attention
// workgroup = 4 waves = 64 queries of one (b,h); K/V tiles staged in LDS via
// the Tensor Data Mover; each wave owns 16 query rows.
__global__ __launch_bounds__(128) void k_attn(
    const bf16* __restrict__ qb, const bf16* __restrict__ kb,
    const bf16* __restrict__ vT, const float* __restrict__ qf,
    float* __restrict__ out)
{
  __shared__ __align__(128) bf16 ldsK[32 * 64];     // [key][d]   row stride 64
  __shared__ __align__(128) bf16 ldsV[64 * 32];     // [d][key]   row stride 32
  __shared__ __align__(128) bf16 ldsP[4][16 * 32];  // per-wave P relayout
  const int wslot = threadIdx.x >> 5;
  const int lane  = threadIdx.x & 31;
  const int lo = lane & 15, hi = lane >> 4;
  const int blk = blockIdx.x;          // 512 blocks
  const int b   = blk >> 7;            // 128 blocks per batch
  const int rem = blk & 127;
  const int h   = rem >> 5;            // 32 blocks per head
  const int q0  = ((rem & 31) << 6) + (wslot << 4);
  const int hd0 = h * 64;

  // Q A-fragments for the whole K loop: A[m=query, k=d], d in [0,64)
  AFrag qa[2];
  const bf16* qrow = qb + ((size_t)(b * 2048 + q0 + lo)) * 256 + hd0;
#pragma unroll
  for (int i = 0; i < 2; ++i) {
    qa[i].h[0] = *(const v8bf*)(qrow + i * 32 + hi * 16);
    qa[i].h[1] = *(const v8bf*)(qrow + i * 32 + hi * 16 + 8);
  }

  const v8f vzero = {};
  v8f o[4]; v8f mrow, lrow;
#pragma unroll
  for (int t = 0; t < 4; ++t) o[t] = vzero;
#pragma unroll
  for (int r = 0; r < 8; ++r) { mrow[r] = -3.0e38f; lrow[r] = 0.0f; }

  const float scale = 0.0625f;   // 1/sqrt(256)
  bf16* pl = ldsP[wslot];
#ifdef USE_TDM
  const unsigned long long kBase =
      (unsigned long long)(const void*)(kb) +
      ((unsigned long long)(b * 2048) * 256 + hd0) * 2ull;
  const unsigned long long vBase =
      (unsigned long long)(const void*)(vT) +
      ((unsigned long long)b * (256 * 2048) + (unsigned long long)hd0 * 2048) * 2ull;
  const unsigned ldsK_off = (unsigned)(unsigned long long)(const void*)&ldsK[0];
  const unsigned ldsV_off = (unsigned)(unsigned long long)(const void*)&ldsV[0];
#endif

  for (int k0 = 0; k0 < 2048; k0 += 32) {
#ifdef USE_TDM
    if (threadIdx.x < 32) {            // wave 0 drives the Tensor Data Mover
      tdm_load_2d(ldsK_off, kBase + (unsigned long long)k0 * 512ull, 64, 32, 256);
      tdm_load_2d(ldsV_off, vBase + (unsigned long long)k0 * 2ull,   32, 64, 2048);
      __builtin_amdgcn_s_wait_tensorcnt(0);
    }
#else
    for (int idx = threadIdx.x; idx < 2048; idx += 128) {
      const int key = idx >> 6, d = idx & 63;
      ldsK[idx] = kb[((size_t)(b * 2048 + k0 + key)) * 256 + hd0 + d];
      const int dv = idx >> 5, kv = idx & 31;
      ldsV[idx] = vT[(size_t)b * (256 * 2048) + (size_t)(hd0 + dv) * 2048 + k0 + kv];
    }
#endif
    __syncthreads();                   // tile visible to all 4 waves

    // S tiles: s0 = keys 0..15 of block, s1 = keys 16..31
    v8f s0 = vzero, s1 = vzero;
#pragma unroll
    for (int i = 0; i < 2; ++i) {
      v16bf b0 = *(const v16bf*)(ldsK + lo * 64 + i * 32 + hi * 16);
      v16bf b1 = *(const v16bf*)(ldsK + (16 + lo) * 64 + i * 32 + hi * 16);
      s0 = wmma_bf16(qa[i].v, b0, s0);
      s1 = wmma_bf16(qa[i].v, b1, s1);
    }

    // online softmax: row stats live across the 16-lane half (row = 8*hi + r)
    v8f mnew;
#pragma unroll
    for (int r = 0; r < 8; ++r) {
      s0[r] *= scale; s1[r] *= scale;
      mnew[r] = fmaxf(s0[r], s1[r]);
    }
#pragma unroll
    for (int off = 1; off < 16; off <<= 1)
#pragma unroll
      for (int r = 0; r < 8; ++r)
        mnew[r] = fmaxf(mnew[r], __shfl_xor(mnew[r], off, 32));

#pragma unroll
    for (int r = 0; r < 8; ++r) {
      mnew[r] = fmaxf(mnew[r], mrow[r]);
      const float corr = __expf(mrow[r] - mnew[r]);
      mrow[r] = mnew[r];
      lrow[r] *= corr;
      o[0][r] *= corr; o[1][r] *= corr; o[2][r] *= corr; o[3][r] *= corr;
      const float p0 = __expf(s0[r] - mnew[r]);
      const float p1 = __expf(s1[r] - mnew[r]);
      lrow[r] += p0 + p1;              // per-lane partial; reduced at end
      const int row = hi * 8 + r;      // C-layout -> LDS row-major 16x32
      pl[row * 32 + lo]      = (bf16)p0;
      pl[row * 32 + 16 + lo] = (bf16)p1;
    }
    __asm volatile("s_wait_dscnt 0" ::: "memory");

    // re-read P in A-fragment layout
    AFrag pa;
    pa.h[0] = *(const v8bf*)(pl + lo * 32 + hi * 16);
    pa.h[1] = *(const v8bf*)(pl + lo * 32 + hi * 16 + 8);

    // O += P(16x32) * V(32x64), V B-fragment contiguous along keys in ldsV
#pragma unroll
    for (int t = 0; t < 4; ++t) {
      v16bf bv = *(const v16bf*)(ldsV + (t * 16 + lo) * 32 + hi * 16);
      o[t] = wmma_bf16(pa.v, bv, o[t]);
    }
    __syncthreads();                   // all waves done with tile before reuse
  }

  // final row-sum reduction and epilogue (residual with projected q)
#pragma unroll
  for (int off = 1; off < 16; off <<= 1)
#pragma unroll
    for (int r = 0; r < 8; ++r)
      lrow[r] += __shfl_xor(lrow[r], off, 32);

#pragma unroll
  for (int r = 0; r < 8; ++r) {
    const int row = q0 + hi * 8 + r;
    const float inv = 1.0f / lrow[r];
    const size_t base = ((size_t)(b * 2048 + row)) * 256 + hd0;
#pragma unroll
    for (int t = 0; t < 4; ++t) {
      const size_t idx = base + t * 16 + lo;
      out[idx] = qf[idx] + o[t][r] * inv;
    }
  }
}

// ------------------------------------------------------------------ layernorm
__global__ __launch_bounds__(256) void k_ln(
    const float* __restrict__ x, const float* __restrict__ g,
    const float* __restrict__ beta, float* __restrict__ outf, int nrows)
{
  const int row = blockIdx.x * 8 + (threadIdx.x >> 5);
  if (row >= nrows) return;
  const int lane = threadIdx.x & 31;
  const float* xr = x + (size_t)row * 256;
  float v[8];
  float s = 0.f, s2 = 0.f;
#pragma unroll
  for (int i = 0; i < 8; ++i) {
    v[i] = xr[lane * 8 + i];
    s += v[i]; s2 += v[i] * v[i];
  }
#pragma unroll
  for (int off = 1; off < 32; off <<= 1) {
    s  += __shfl_xor(s, off, 32);
    s2 += __shfl_xor(s2, off, 32);
  }
  const float mu  = s * (1.0f / 256.0f);
  const float var = s2 * (1.0f / 256.0f) - mu * mu;
  const float rs  = rsqrtf(var + 1e-5f);
#pragma unroll
  for (int i = 0; i < 8; ++i) {
    const int c = lane * 8 + i;
    outf[(size_t)row * 256 + c] = (v[i] - mu) * rs * g[c] + beta[c];
  }
}

// -------------------------------------------------------------------- launch
extern "C" void kernel_launch(void* const* d_in, const int* in_sizes, int n_in,
                              void* d_out, int out_size, void* d_ws, size_t ws_size,
                              hipStream_t stream) {
  const float* Q  = (const float*)d_in[0];
  const float* K  = (const float*)d_in[1];
  const float* Wq = (const float*)d_in[2];
  const float* bq = (const float*)d_in[3];
  const float* Wk = (const float*)d_in[4];
  const float* bk = (const float*)d_in[5];
  const float* Wv = (const float*)d_in[6];
  const float* bv = (const float*)d_in[7];
  const float* Wo = (const float*)d_in[8];
  const float* bo = (const float*)d_in[9];
  const float* g0 = (const float*)d_in[10];
  const float* b0 = (const float*)d_in[11];
  const float* g1 = (const float*)d_in[12];
  const float* b1 = (const float*)d_in[13];

  char* ws = (char*)d_ws;
  const size_t MB = 1u << 20;
  bf16*  Wqb  = (bf16*)(ws + 0);         // 128 KB each
  bf16*  Wkb  = Wqb + 65536;
  bf16*  Wvb  = Wkb + 65536;
  bf16*  Wob  = Wvb + 65536;
  float* qf32 = (float*)(ws + 1 * MB);   // 8 MB  q proj fp32 (residual)
  bf16*  qbb  = (bf16*)(ws + 9 * MB);    // 4 MB  q proj bf16
  bf16*  kbb  = (bf16*)(ws + 13 * MB);   // 4 MB  k proj bf16 (key,d)
  bf16*  vTb  = (bf16*)(ws + 17 * MB);   // 4 MB  v proj bf16 (b,d,key)
  float* Oat  = (float*)(ws + 21 * MB);  // 8 MB  attention out fp32
  float* L0f  = (float*)(ws + 29 * MB);  // 8 MB  LN0 fp32
  float* Ff   = (float*)(ws + 37 * MB);  // 8 MB  FFN+residual fp32

  k_cvt<<<256, 256, 0, stream>>>(Wq, Wqb, 65536);
  k_cvt<<<256, 256, 0, stream>>>(Wk, Wkb, 65536);
  k_cvt<<<256, 256, 0, stream>>>(Wv, Wvb, 65536);
  k_cvt<<<256, 256, 0, stream>>>(Wo, Wob, 65536);

  // 2048 waves = 512 m-tiles x 4 column strips; 4 waves/block
  k_gemm<<<512, 128, 0, stream>>>(Q, Wqb, bq, qf32, qbb, nullptr, 0);
  k_gemm<<<512, 128, 0, stream>>>(K, Wkb, bk, nullptr, kbb, nullptr, 1);
  k_gemm<<<512, 128, 0, stream>>>(K, Wvb, bv, nullptr, vTb, nullptr, 2);

  // 512 blocks = 4 batch * 4 heads * 32 query-superblocks (64 q each)
  k_attn<<<512, 128, 0, stream>>>(qbb, kbb, vTb, qf32, Oat);

  k_ln<<<1024, 256, 0, stream>>>(Oat, g0, b0, L0f, 8192);
  k_gemm<<<512, 128, 0, stream>>>(L0f, Wob, bo, Ff, nullptr, L0f, 3);
  k_ln<<<1024, 256, 0, stream>>>(Ff, g1, b1, (float*)d_out, 8192);
}
